// QJLBitPacked_42700564857504
// MI455X (gfx1250) — matrix operand order repaired and checked by reading, not compiled, not run on hardware
//
#include <hip/hip_runtime.h>
#include <math.h>

// QJL sign-sketch inner-product estimator:
// out[r] = sqrt(pi/2)/256 * sum_p (q[r]·S[p]) * sign(k[r]·S[p])
// Two M=262144, N=256, K=128 GEMMs fused, run on v_wmma_f32_16x16x32_bf16.
// Fragments gathered from LDS with 16B-aligned ds_load_b128.

typedef __attribute__((ext_vector_type(16))) __bf16    v16bf;
typedef __attribute__((ext_vector_type(8)))  float     v8f;
typedef __attribute__((ext_vector_type(4)))  unsigned  v4u;

#define DIMK 128
#define NPROJ 256
#define ROWS_PER_BLOCK 128
#define STR 136   // padded LDS row stride (bf16 elems): 272B = 17*16B -> keeps 16B align,
                  // row-to-row bank advance of 4 -> conflict-free b128 gathers

union FragA { v16bf v; v4u u4[2]; };

__global__ __launch_bounds__(256)
void qjl_wmma_kernel(const float* __restrict__ q,
                     const float* __restrict__ k,
                     const float* __restrict__ S,
                     float* __restrict__ out,
                     int totalRows)
{
    extern __shared__ __bf16 lds[];
    __bf16* s_lds = lds;                          // NPROJ * STR
    __bf16* q_lds = s_lds + NPROJ * STR;          // ROWS_PER_BLOCK * STR
    __bf16* k_lds = q_lds + ROWS_PER_BLOCK * STR; // ROWS_PER_BLOCK * STR

    const int tid     = threadIdx.x;
    const int rowBase = blockIdx.x * ROWS_PER_BLOCK;

    // ---- Stage S (256x128 fp32 -> bf16), coalesced float4 reads ----
    for (int i = tid; i < NPROJ * (DIMK / 4); i += 256) {
        const int r  = i >> 5;        // / (DIMK/4)
        const int c4 = i & 31;
        const float4 f = ((const float4*)S)[i];
        __bf16* d = &s_lds[r * STR + c4 * 4];
        d[0] = (__bf16)f.x; d[1] = (__bf16)f.y; d[2] = (__bf16)f.z; d[3] = (__bf16)f.w;
    }
    // ---- Stage this block's 128-row q/k tiles ----
    for (int i = tid; i < ROWS_PER_BLOCK * (DIMK / 4); i += 256) {
        const int r  = i >> 5;
        const int c4 = i & 31;
        int gr = rowBase + r; if (gr >= totalRows) gr = totalRows - 1;
        const long gidx = (long)gr * (DIMK / 4) + c4;
        const float4 fq = ((const float4*)q)[gidx];
        const float4 fk = ((const float4*)k)[gidx];
        __bf16* dq = &q_lds[r * STR + c4 * 4];
        dq[0] = (__bf16)fq.x; dq[1] = (__bf16)fq.y; dq[2] = (__bf16)fq.z; dq[3] = (__bf16)fq.w;
        __bf16* dk = &k_lds[r * STR + c4 * 4];
        dk[0] = (__bf16)fk.x; dk[1] = (__bf16)fk.y; dk[2] = (__bf16)fk.z; dk[3] = (__bf16)fk.w;
    }
    __syncthreads();

    const int lane     = tid & 31;
    const int wave     = tid >> 5;
    const int g        = lane >> 4;   // half-wave (selects K sub-group in A/B layouts)
    const int m        = lane & 15;   // A: row in tile; B: column
    const int rowInBlk = wave * 16;

    // ---- Build A fragments (16x32 bf16 layout):
    //      elements 0..7  -> K = kb + g*8 .. +7   (16B contiguous, aligned)
    //      elements 8..15 -> K = kb + 16 + g*8 .. (16B contiguous, aligned)
    FragA aq[4], ak[4];
    const __bf16* qrow = &q_lds[(rowInBlk + m) * STR];
    const __bf16* krow = &k_lds[(rowInBlk + m) * STR];
    #pragma unroll
    for (int kk = 0; kk < 4; ++kk) {
        const int kb = kk * 32 + g * 8;
        aq[kk].u4[0] = *(const v4u*)&qrow[kb];
        aq[kk].u4[1] = *(const v4u*)&qrow[kb + 16];
        ak[kk].u4[0] = *(const v4u*)&krow[kb];
        ak[kk].u4[1] = *(const v4u*)&krow[kb + 16];
    }

    float comb[8];
    #pragma unroll
    for (int r = 0; r < 8; ++r) comb[r] = 0.0f;

    // ---- 16 projection tiles of 16 columns each; running LDS pointer, unroll x2 so
    //      the next tile's DS loads / WMMAs overlap the previous tile's sign-combine ----
    const __bf16* srow = &s_lds[m * STR + g * 8];
    #pragma unroll 2
    for (int pt = 0; pt < 16; ++pt) {
        // B fragment (32x16 bf16 layout): lane m = column N, elements 0..7 -> K = kb + g*8 + h,
        // elements 8..15 -> K = kb + 16 + g*8 + (h-8) -> two aligned ds_load_b128
        FragA b[4];
        #pragma unroll
        for (int kk = 0; kk < 4; ++kk) {
            b[kk].u4[0] = *(const v4u*)&srow[kk * 32];
            b[kk].u4[1] = *(const v4u*)&srow[kk * 32 + 16];
        }
        srow += 16 * STR;

        v8f accQ = {}; v8f accK = {};
        #pragma unroll
        for (int kk = 0; kk < 4; ++kk) {
            accQ = __builtin_amdgcn_wmma_f32_16x16x32_bf16(
                false, aq[kk].v, false, b[kk].v, (short)0, accQ, false, false);
            accK = __builtin_amdgcn_wmma_f32_16x16x32_bf16(
                false, ak[kk].v, false, b[kk].v, (short)0, accK, false, false);
        }
        // C/D layout: same lane/VGPR holds matching (M,N) of accQ/accK -> pure VALU combine
        #pragma unroll
        for (int r = 0; r < 8; ++r)
            comb[r] += (accK[r] > 0.0f) ? accQ[r] : -accQ[r];
    }

    // ---- Reduce over N (16 lanes per half-wave), write M=r (lane 0) / M=r+8 (lane 16) ----
    const float scale = 1.25331413731550025f / 256.0f; // sqrt(pi/2)/NUM_PROJ
    #pragma unroll
    for (int r = 0; r < 8; ++r) {
        float v = comb[r];
        v += __shfl_xor(v, 8, 32);
        v += __shfl_xor(v, 4, 32);
        v += __shfl_xor(v, 2, 32);
        v += __shfl_xor(v, 1, 32);
        const int orow = rowBase + rowInBlk + g * 8 + r;
        if (m == 0 && orow < totalRows) out[orow] = scale * v;
    }
}

extern "C" void kernel_launch(void* const* d_in, const int* in_sizes, int n_in,
                              void* d_out, int out_size, void* d_ws, size_t ws_size,
                              hipStream_t stream) {
    const float* q = (const float*)d_in[0];  // (2,32,4096,128) fp32
    const float* k = (const float*)d_in[1];  // (2,32,4096,128) fp32
    const float* S = (const float*)d_in[2];  // (256,128) fp32
    float* out = (float*)d_out;              // (2,32,4096) fp32

    const int totalRows = in_sizes[0] / DIMK;                 // 262144
    const int blocks = (totalRows + ROWS_PER_BLOCK - 1) / ROWS_PER_BLOCK;
    const size_t ldsBytes =
        (size_t)(NPROJ * STR + 2 * ROWS_PER_BLOCK * STR) * sizeof(__bf16); // ~136 KB

    qjl_wmma_kernel<<<blocks, 256, ldsBytes, stream>>>(q, k, S, out, totalRows);
}
